// SpatiallyAdaptiveSirenLayer_31817117728925
// MI455X (gfx1250) — matrix operand order repaired and checked by reading, not compiled, not run on hardware
//
#include <hip/hip_runtime.h>
#include <hip/hip_bf16.h>
#include <math.h>

typedef __attribute__((ext_vector_type(16))) _Float16 v16h;
typedef __attribute__((ext_vector_type(8)))  _Float16 v8h;
typedef __attribute__((ext_vector_type(8)))  float    v8f;

#define SIREN_B    2
#define SIREN_N    16384
#define SIREN_M    1024
#define SIREN_IN   256
#define SIREN_OUT  256
#define OMEGA0     30.0f
#define EPS_D2     1e-4f

// ---------------------------------------------------------------------------
// Kernel 1: per query point, compute
//   omega[p] = 30 * (1 + clamp(freq_net(qc), 0, 5) * exp(-min_dist(qc, atoms)))
// One thread per point. Atoms for the batch staged SoA in LDS (12 KB).
// ---------------------------------------------------------------------------
__global__ void __launch_bounds__(256)
siren_omega_kernel(const float* __restrict__ qc,   // [B, N, 3]
                   const float* __restrict__ ac,   // [B, M, 3]
                   const float* __restrict__ fw1,  // [16, 3]
                   const float* __restrict__ fb1,  // [16]
                   const float* __restrict__ fw2,  // [1, 16]
                   const float* __restrict__ fb2,  // [1]
                   float* __restrict__ omega)      // [B*N]
{
    __shared__ float ax[SIREN_M];
    __shared__ float ay[SIREN_M];
    __shared__ float az[SIREN_M];

    const int batch = blockIdx.y;
    const int tid   = threadIdx.x;

    const float* acb = ac + (size_t)batch * SIREN_M * 3;
    for (int i = tid; i < SIREN_M; i += 256) {
        ax[i] = acb[i * 3 + 0];
        ay[i] = acb[i * 3 + 1];
        az[i] = acb[i * 3 + 2];
    }
    __syncthreads();

    const int p = batch * SIREN_N + blockIdx.x * 256 + tid;
    const float qx = qc[(size_t)p * 3 + 0];
    const float qy = qc[(size_t)p * 3 + 1];
    const float qz = qc[(size_t)p * 3 + 2];

    // min squared distance over atoms
    float d2min = 3.4e38f;
#pragma unroll 4
    for (int m = 0; m < SIREN_M; ++m) {
        float dx = qx - ax[m];
        float dy = qy - ay[m];
        float dz = qz - az[m];
        float d2 = fmaf(dx, dx, fmaf(dy, dy, dz * dz));
        d2min = fminf(d2min, d2);
    }
    float min_dist = sqrtf(fmaxf(d2min, EPS_D2));

    // freq_net: Linear(3,16) -> softplus -> Linear(16,1)
    float ls = fb2[0];
#pragma unroll
    for (int j = 0; j < 16; ++j) {
        float pre = fb1[j] + qx * fw1[j * 3 + 0] + qy * fw1[j * 3 + 1] + qz * fw1[j * 3 + 2];
        float h   = (pre > 20.0f) ? pre : log1pf(expf(pre));
        ls = fmaf(fw2[j], h, ls);
    }
    // clamp [0,5]; fmin/fmax drop NaN -> matches nan_to_num(clip(...), 0)
    ls = fminf(fmaxf(ls, 0.0f), 5.0f);
    if (isnan(ls)) ls = 0.0f;

    omega[p] = OMEGA0 * (1.0f + ls * expf(-min_dist));
}

// ---------------------------------------------------------------------------
// Kernel 2: out[r, c] = sin(omega[r] * (sum_k x[r,k] * W[c,k] + b[c]))
// 8 waves/block; block tile = 256 rows x 32 cols. Each wave owns a 32x32
// strip: 2 A-fragments x 2 B-tiles -> 4 independent f32 accumulators
// (32 VGPRs). Each LDS B-fragment feeds 2 WMMAs (1 ds_load_b128 per WMMA)
// and each A-fragment feeds 2; 4 independent V_WMMA_F32_16X16X32_F16 per
// K-chunk cover the WMMA hazards. This shape keeps peak live VGPRs at the
// level that allocated spill-free (the 8-accumulator variant forced the
// allocator to spill accumulators to scratch every chunk).
// W col-tile (32 x 256) staged as f16 in LDS with padded rows so the
// per-column 4-bank skew makes ds_load_b128 B-fragment reads conflict-free.
// ---------------------------------------------------------------------------
#define CPB      32    // columns per block (2 x 16-col B tiles)
#define RPW      32    // rows per wave    (2 x 16-row A fragments)
#define WROW_PAD 264   // 256 halves + 8 pad -> 528B row stride (16B aligned)

static __device__ __forceinline__ v16h cvt_a_frag(const float* xrow, int k0, int off) {
    const float4 fa0 = *(const float4*)(xrow + k0 + off);
    const float4 fa1 = *(const float4*)(xrow + k0 + off + 4);
    const float4 fa2 = *(const float4*)(xrow + k0 + 16 + off);
    const float4 fa3 = *(const float4*)(xrow + k0 + 16 + off + 4);
    v16h a;
    a[0]  = (_Float16)fa0.x; a[1]  = (_Float16)fa0.y;
    a[2]  = (_Float16)fa0.z; a[3]  = (_Float16)fa0.w;
    a[4]  = (_Float16)fa1.x; a[5]  = (_Float16)fa1.y;
    a[6]  = (_Float16)fa1.z; a[7]  = (_Float16)fa1.w;
    a[8]  = (_Float16)fa2.x; a[9]  = (_Float16)fa2.y;
    a[10] = (_Float16)fa2.z; a[11] = (_Float16)fa2.w;
    a[12] = (_Float16)fa3.x; a[13] = (_Float16)fa3.y;
    a[14] = (_Float16)fa3.z; a[15] = (_Float16)fa3.w;
    return a;
}

__global__ void __launch_bounds__(256, 1)
siren_wmma_kernel(const float* __restrict__ x,     // [B*N, 256]
                  const float* __restrict__ W,     // [256, 256]
                  const float* __restrict__ bias,  // [256]
                  const float* __restrict__ omega, // [B*N]
                  float* __restrict__ out)         // [B*N, 256]
{
    __shared__ _Float16 Wh[CPB * WROW_PAD];  // 32 cols x 256 K (f16), ~17 KB

    const int tid     = threadIdx.x;
    const int lane    = tid & 31;
    const int wave    = tid >> 5;
    const int colBase = blockIdx.x * CPB;
    const int rowBase = blockIdx.y * (8 * RPW) + wave * RPW;

    // Stage W tile for this column block: convert f32 -> f16 into LDS.
    for (int idx = tid; idx < CPB * SIREN_IN; idx += 256) {
        const int col = idx >> 8;        // 0..31
        const int k   = idx & 255;
        Wh[col * WROW_PAD + k] = (_Float16)W[(size_t)(colBase + col) * SIREN_IN + k];
    }
    __syncthreads();

    // Per-lane fragment geometry (ISA 16-bit A 16x32 / B 32x16 wave32 maps)
    const int half32 = lane >> 4;          // 0: lanes 0-15, 1: lanes 16-31
    const int bcol   = lane & 15;
    const float* xrow0 = x + (size_t)(rowBase + (lane & 15)) * SIREN_IN;
    const float* xrow1 = xrow0 + (size_t)16 * SIREN_IN;

    v8f acc[2][2] = {};
#pragma unroll
    for (int kc = 0; kc < SIREN_IN / 32; ++kc) {
        const int k0  = kc * 32;
        const int off = half32 ? 8 : 0;

        if (kc + 1 < SIREN_IN / 32)
            __builtin_prefetch(xrow0 + k0 + 32 + off, 0, 3);

        // Two A fragments: rows [rowBase, +16) and [rowBase+16, +32)
        const v16h a0 = cvt_a_frag(xrow0, k0, off);
        const v16h a1 = cvt_a_frag(xrow1, k0, off);

        // 2 column tiles; each B fragment feeds both A fragments
        const int koff = k0 + (half32 ? 8 : 0);
#pragma unroll
        for (int t = 0; t < 2; ++t) {
            // B fragment: B[k][n] = W[n][k]; lane n holds K koff..+7 and koff+16..+23
            const int col = t * 16 + bcol;
            const v8h* bp = (const v8h*)(&Wh[col * WROW_PAD + koff]);
            const v8h b0 = bp[0];
            const v8h b1 = bp[2];
            v16h bfrag;
#pragma unroll
            for (int i = 0; i < 8; ++i) { bfrag[i] = b0[i]; bfrag[8 + i] = b1[i]; }

            acc[0][t] = __builtin_amdgcn_wmma_f32_16x16x32_f16(
                false, a0, false, bfrag, (short)0, acc[0][t], false, false);
            acc[1][t] = __builtin_amdgcn_wmma_f32_16x16x32_f16(
                false, a1, false, bfrag, (short)0, acc[1][t], false, false);
        }
    }

    // Epilogue: C/D map -> VGPR j holds M=j (lanes 0-15) / M=j+8 (lanes 16-31)
    float bc[2];
#pragma unroll
    for (int t = 0; t < 2; ++t) bc[t] = bias[colBase + t * 16 + bcol];

#pragma unroll
    for (int s = 0; s < 2; ++s) {
#pragma unroll
        for (int j = 0; j < 8; ++j) {
            const int r  = rowBase + s * 16 + j + (half32 ? 8 : 0);
            const float om = omega[r];
            float* orow = out + (size_t)r * SIREN_OUT + colBase + bcol;
#pragma unroll
            for (int t = 0; t < 2; ++t) {
                const float pre = acc[s][t][j] + bc[t];
                orow[t * 16] = sinf(om * pre);
            }
        }
    }
}

// ---------------------------------------------------------------------------
extern "C" void kernel_launch(void* const* d_in, const int* in_sizes, int n_in,
                              void* d_out, int out_size, void* d_ws, size_t ws_size,
                              hipStream_t stream) {
    const float* x   = (const float*)d_in[0];  // [2,16384,256]
    const float* qc  = (const float*)d_in[1];  // [2,16384,3]
    const float* ac  = (const float*)d_in[2];  // [2,1024,3]
    const float* W   = (const float*)d_in[3];  // [256,256]
    const float* b   = (const float*)d_in[4];  // [256]
    const float* fw1 = (const float*)d_in[5];  // [16,3]
    const float* fb1 = (const float*)d_in[6];  // [16]
    const float* fw2 = (const float*)d_in[7];  // [1,16]
    const float* fb2 = (const float*)d_in[8];  // [1]
    float* out       = (float*)d_out;

    float* omega = (float*)d_ws;               // B*N floats = 128 KB scratch

    dim3 g1(SIREN_N / 256, SIREN_B);
    siren_omega_kernel<<<g1, 256, 0, stream>>>(qc, ac, fw1, fb1, fw2, fb2, omega);

    dim3 g2(SIREN_OUT / CPB, (SIREN_B * SIREN_N) / (8 * RPW));
    siren_wmma_kernel<<<g2, 256, 0, stream>>>(x, W, b, omega, out);
}
